// InteractionNetwork_54778012893568
// MI455X (gfx1250) — compile-verified, matrix operand order checked by reading, not run on hardware
//
#include <hip/hip_runtime.h>

typedef __attribute__((ext_vector_type(16))) __bf16 bf16x16;
typedef __attribute__((ext_vector_type(2)))  __bf16 bf16x2;
typedef __attribute__((ext_vector_type(8)))  float  f32x8;
typedef __attribute__((ext_vector_type(2)))  float  f32x2;

#define NN 1024

union AFrag {
  bf16x16 v;
  __bf16  h[16];
  unsigned u[8];
  uint4   q[2];
};

// packed conversion -> v_cvt_pk_bf16_f32 (RTNE), used for both init and hot paths
__device__ __forceinline__ unsigned pack_bf16(float a, float b) {
  f32x2 v = {a, b};
  bf16x2 r = __builtin_convertvector(v, bf16x2);
  return __builtin_bit_cast(unsigned, r);
}

// single-instruction relu; inline asm so clang cannot re-expand it to canonicalize+max
__device__ __forceinline__ float relu(float x) {
  float r;
  asm("v_max_num_f32 %0, %1, 0" : "=v"(r) : "v"(x));
  return r;
}

__global__ void zero_f32(float* __restrict__ p, int n) {
  int i = blockIdx.x * 256 + threadIdx.x;
  if (i < n) p[i] = 0.0f;
}

__global__ __launch_bounds__(256) void edge_mlp(
    const float* __restrict__ pos, const float* __restrict__ W1,
    const float* __restrict__ b1,  const float* __restrict__ W2,
    const float* __restrict__ b2,  const float* __restrict__ W3,
    float* __restrict__ out)
{
  // per-wave activation staging (h1, then reused for h2): [wave][m=16][64 pos pad 72]
  // K-position permutation: hidden unit (16t + q) lives at position 4q + t.
  __shared__ __bf16 stage[8][16][72];

  const int lane = threadIdx.x & 31;
  const int wave = threadIdx.x >> 5;
  const int nlo  = lane & 15;
  const int ksel = lane >> 4;

  const int blk = blockIdx.x;
  const int jg = blk & 7;
  const int it = (blk >> 3) & 63;
  const int b  = blk >> 9;
  const int i_base = it * 16;
  const int j_base = (jg * 8 + wave) * 16;

  // ---- W2 fragments (bf16 B layout) with the K-permutation folded in ----
  AFrag w2f[2][4];
#pragma unroll
  for (int ks = 0; ks < 2; ++ks)
#pragma unroll
    for (int t = 0; t < 4; ++t) {
      const int n = t * 16 + nlo;
#pragma unroll
      for (int r = 0; r < 8; ++r) {
        const int k0 = 2 * (r & 3) + 8 * ksel + 16 * (r >> 2) + 32 * ks;
        const int hu0 = 16 * (k0 & 3) + (k0 >> 2);            // pi^-1(k0)
        const int hu1 = 16 * ((k0 + 1) & 3) + ((k0 + 1) >> 2);
        w2f[ks][t].u[r] = pack_bf16(W2[hu0 * 64 + n], W2[hu1 * 64 + n]);
      }
    }

  // ---- W1 fragments: K 0..6 = W1 rows, K 7 = b1 (bias via constant-1 feature) ----
  AFrag w1f[4];
#pragma unroll
  for (int t = 0; t < 4; ++t) {
    const int n = t * 16 + nlo;
#pragma unroll
    for (int r = 0; r < 8; ++r) {
      const int k0 = 2 * (r & 3) + 8 * ksel + 16 * (r >> 2);
      const int k1 = k0 + 1;
      const float v0 = (k0 < 7) ? W1[k0 * 64 + n] : ((k0 == 7) ? b1[n] : 0.0f);
      const float v1 = (k1 < 7) ? W1[k1 * 64 + n] : ((k1 == 7) ? b1[n] : 0.0f);
      w1f[t].u[r] = pack_bf16(v0, v1);
    }
  }

  // ---- W3 fragments (64x16, cols 0..2 = W3, rest zero), same K-permutation ----
  AFrag w3f[2];
#pragma unroll
  for (int ks = 0; ks < 2; ++ks)
#pragma unroll
    for (int r = 0; r < 8; ++r) {
      const int k0 = 2 * (r & 3) + 8 * ksel + 16 * (r >> 2) + 32 * ks;
      const int hu0 = 16 * (k0 & 3) + (k0 >> 2);
      const int hu1 = 16 * ((k0 + 1) & 3) + ((k0 + 1) >> 2);
      const float v0 = (nlo < 3) ? W3[hu0 * 3 + nlo] : 0.0f;
      const float v1 = (nlo < 3) ? W3[hu1 * 3 + nlo] : 0.0f;
      w3f[ks].u[r] = pack_bf16(v0, v1);
    }

  float b2v[4];
#pragma unroll
  for (int t = 0; t < 4; ++t) b2v[t] = b2[t * 16 + nlo];
  // b3 and the diagonal cancel exactly under antisymmetrization.

  const float pjx = pos[(b * NN + j_base + nlo) * 3 + 0];
  const float pjy = pos[(b * NN + j_base + nlo) * 3 + 1];
  const float pjz = pos[(b * NN + j_base + nlo) * 3 + 2];

  // column-sum accumulator: lane nlo = component c (valid for nlo < 3), j = j_base + r + 8*ksel
  float colacc[8];
#pragma unroll
  for (int r = 0; r < 8; ++r) colacc[r] = 0.f;

  for (int chunk = 0; chunk < 16; ++chunk) {
    const int i = i_base + chunk;
    const float pix = pos[(b * NN + i) * 3 + 0];
    const float piy = pos[(b * NN + i) * 3 + 1];
    const float piz = pos[(b * NN + i) * 3 + 2];
    const float dx = pjx - pix, dy = pjy - piy, dz = pjz - piz;
    const float dsq = dx * dx + dy * dy + dz * dz + 1.0e-4f;   // EPS^2
    const float rin = rsqrtf(dsq);
    const float inv3 = rin * rin * rin;

    // A fragment: M = 16 edges (j), K = [dx,dy,dz,fx,fy,fz,dsq,1] padded to 32
    AFrag af;
    af.q[0] = make_uint4(0u, 0u, 0u, 0u);
    af.q[1] = make_uint4(0u, 0u, 0u, 0u);
    if (ksel == 0) {
      af.u[0] = pack_bf16(dx, dy);
      af.u[1] = pack_bf16(dz, dx * inv3);
      af.u[2] = pack_bf16(dy * inv3, dz * inv3);
      af.u[3] = pack_bf16(dsq, 1.0f);
    }

    // -------- layer 1: h1 = relu(e @ W1 + b1) (bias inside WMMA) --------
    const f32x8 zacc = {};
    f32x8 c1[4];
#pragma unroll
    for (int t = 0; t < 4; ++t)
      c1[t] = __builtin_amdgcn_wmma_f32_16x16x32_bf16(false, af.v, false, w1f[t].v,
                                                      (short)0, zacc, false, false);

    // relu -> packed bf16 -> one b64 store per r (permuted K layout)
#pragma unroll
    for (int r = 0; r < 8; ++r) {
      uint2 st;
      st.x = pack_bf16(relu(c1[0][r]), relu(c1[1][r]));
      st.y = pack_bf16(relu(c1[2][r]), relu(c1[3][r]));
      *(uint2*)&stage[wave][r + 8 * ksel][4 * nlo] = st;
    }
    asm volatile("s_wait_dscnt 0x0" ::: "memory");   // same-wave LDS store->load

    // -------- layer 2: h2 = h1 @ W2 + b2 (bias-seeded accumulators) --------
    f32x8 c2[4];
#pragma unroll
    for (int t = 0; t < 4; ++t) {
      f32x8 s;
#pragma unroll
      for (int r = 0; r < 8; ++r) s[r] = b2v[t];
      c2[t] = s;
    }
#pragma unroll
    for (int ks = 0; ks < 2; ++ks) {
      AFrag a;
      a.q[0] = *(const uint4*)&stage[wave][nlo][8 * ksel + 32 * ks];
      a.q[1] = *(const uint4*)&stage[wave][nlo][8 * ksel + 32 * ks + 16];
#pragma unroll
      for (int t = 0; t < 4; ++t)
        c2[t] = __builtin_amdgcn_wmma_f32_16x16x32_bf16(false, a.v, false, w2f[ks][t].v,
                                                        (short)0, c2[t], false, false);
    }

    // relu(h2) -> bf16 -> same staging buffer (safe: LDS is in-order per wave)
#pragma unroll
    for (int r = 0; r < 8; ++r) {
      uint2 st;
      st.x = pack_bf16(relu(c2[0][r]), relu(c2[1][r]));
      st.y = pack_bf16(relu(c2[2][r]), relu(c2[3][r]));
      *(uint2*)&stage[wave][r + 8 * ksel][4 * nlo] = st;
    }
    asm volatile("s_wait_dscnt 0x0" ::: "memory");

    // -------- layer 3 on WMMA: h3 = h2 @ W3pad (D[m][c], c = lane, c < 3 valid) --------
    f32x8 c3 = zacc;
#pragma unroll
    for (int ks = 0; ks < 2; ++ks) {
      AFrag a;
      a.q[0] = *(const uint4*)&stage[wave][nlo][8 * ksel + 32 * ks];
      a.q[1] = *(const uint4*)&stage[wave][nlo][8 * ksel + 32 * ks + 16];
      c3 = __builtin_amdgcn_wmma_f32_16x16x32_bf16(false, a.v, false, w3f[ks].v,
                                                   (short)0, c3, false, false);
    }

    // column sums: per-lane register accumulation, no cross-lane traffic
    float rsum = 0.f;
#pragma unroll
    for (int r = 0; r < 8; ++r) {
      colacc[r] += c3[r];
      rsum += c3[r];
    }
    // row sum for this i: combine the two m-halves, then 3 predicated atomics
    rsum += __shfl_xor(rsum, 16, 32);
    if (lane < 3)
      atomicAdd(&out[(b * NN + i) * 3 + lane], 0.5f * rsum);
  }

  // acc[b,i] -= 0.5*colsum: lane nlo < 3 holds component nlo for j = j_base + r + 8*ksel
  if (nlo < 3) {
#pragma unroll
    for (int r = 0; r < 8; ++r) {
      const int j = j_base + r + 8 * ksel;
      atomicAdd(&out[(b * NN + j) * 3 + nlo], -0.5f * colacc[r]);
    }
  }
}

extern "C" void kernel_launch(void* const* d_in, const int* in_sizes, int n_in,
                              void* d_out, int out_size, void* d_ws, size_t ws_size,
                              hipStream_t stream) {
  (void)n_in; (void)d_ws; (void)ws_size;
  const float* pos = (const float*)d_in[0];
  const float* W1  = (const float*)d_in[1];
  const float* b1  = (const float*)d_in[2];
  const float* W2  = (const float*)d_in[3];
  const float* b2  = (const float*)d_in[4];
  const float* W3  = (const float*)d_in[5];
  // d_in[6] (b3) unused: cancels exactly in the antisymmetrization.
  float* out = (float*)d_out;

  const int B = in_sizes[0] / (NN * 3);           // = 2
  zero_f32<<<(out_size + 255) / 256, 256, 0, stream>>>(out, out_size);
  const int blocks = B * (NN / 16) * ((NN / 16) / 8);  // 1024
  edge_mlp<<<blocks, 256, 0, stream>>>(pos, W1, b1, W2, b2, W3, out);
}
